// DSTP_rnn_66571993089064
// MI455X (gfx1250) — matrix operand order repaired and checked by reading, not compiled, not run on hardware
//
#include <hip/hip_runtime.h>
#include <hip/hip_bf16.h>

typedef __attribute__((ext_vector_type(16))) __bf16 v16bf;
typedef __attribute__((ext_vector_type(8)))  float  v8f;

#define BSZ 16384
#define TM1 9
#define NF  128

__device__ __forceinline__ float sigm(float x) { return 1.0f / (1.0f + __expf(-x)); }

// ---------------- WMMA fragment helpers (CDNA5 wave32 layouts) ----------------
// A fragment: 16x32 bf16 from LDS, row-major with leading dim `lda` (elements).
// lane<16: row m0+lane, K = k0+{0..7} (elems 0..7) and k0+{16..23} (elems 8..15)
// lane>=16: same row set, K base shifted by +8.
__device__ __forceinline__ v16bf load_a_frag(const __hip_bfloat16* sm, int m0, int k0, int lda) {
  int lane = threadIdx.x & 31;
  int m  = m0 + (lane & 15);
  int kb = k0 + ((lane & 16) ? 8 : 0);
  const __hip_bfloat16* p = sm + m * lda + kb;
  union { v16bf v; uint4 q[2]; } u;
  u.q[0] = *(const uint4*)(p);
  u.q[1] = *(const uint4*)(p + 16);
  return u.v;
}

// B fragment: 32x16 bf16, pre-packed in global memory: frag (nt,kt) is a
// contiguous 1 KB blob, lane-major, 16 bf16 per lane.
__device__ __forceinline__ v16bf load_b_frag(const __hip_bfloat16* pk, int kt, int nt, int KT) {
  int lane = threadIdx.x & 31;
  const __hip_bfloat16* p = pk + (((size_t)(nt * KT + kt)) << 9) + lane * 16;
  union { v16bf v; uint4 q[2]; } u;
  u.q[0] = ((const uint4*)p)[0];
  u.q[1] = ((const uint4*)p)[1];
  return u.v;
}

// C/D fragment store: VGPR v holds (M=m0+v | m0+8+v, N = n0 + lane&15)
template <bool BF16OUT>
__device__ __forceinline__ void store_c_frag(void* G, v8f c, int m0, int n0, int ldg) {
  int lane = threadIdx.x & 31;
  int n  = n0 + (lane & 15);
  int mb = m0 + ((lane & 16) ? 8 : 0);
  if constexpr (BF16OUT) {
    __hip_bfloat16* g = (__hip_bfloat16*)G;
#pragma unroll
    for (int v = 0; v < 8; ++v) g[(mb + v) * ldg + n] = __float2bfloat16(c[v]);
  } else {
    float* g = (float*)G;
#pragma unroll
    for (int v = 0; v < 8; ++v) g[(mb + v) * ldg + n] = c[v];
  }
}

// Workgroup-cooperative GEMM over 16x16 output tiles; 8 waves stride the tiles.
template <int KT, bool BF16OUT>
__device__ __forceinline__ void gemm_tiles(const __hip_bfloat16* A, const __hip_bfloat16* pk,
                                           void* G, int MTt, int NTt, int lda, int ldg) {
  int wave  = threadIdx.x >> 5;
  int total = MTt * NTt;
  for (int tile = wave; tile < total; tile += 8) {
    int mt = tile / NTt, nt = tile - mt * NTt;
    v8f acc = {0.f, 0.f, 0.f, 0.f, 0.f, 0.f, 0.f, 0.f};
#pragma unroll
    for (int kt = 0; kt < KT; ++kt) {
      v16bf a = load_a_frag(A, mt * 16, kt * 32, lda);
      v16bf b = load_b_frag(pk, kt, nt, KT);
      acc = __builtin_amdgcn_wmma_f32_16x16x32_bf16(false, a, false, b, (short)0, acc,
                                                    false, false);
    }
    store_c_frag<BF16OUT>(G, acc, mt * 16, nt * 16, ldg);
  }
}

// ---------------- weight packer: B[k][n] -> fragment-layout bf16 blobs ----------------
// transposed=1: B[k][n] = W[n*Kd + k]   (W stored (Nout, K) row-major, i.e. x @ W^T)
// transposed=0: B[k][n] = W[k*Nout + n]
__global__ void pack_b_kernel(const float* __restrict__ W0, const float* __restrict__ W1,
                              int K1, int Ktot, int Nout, int transposed,
                              __hip_bfloat16* __restrict__ out) {
  int idx = blockIdx.x * blockDim.x + threadIdx.x;
  if (idx >= Ktot * Nout) return;
  int j    = idx & 15;
  int lane = (idx >> 4) & 31;
  int frag = idx >> 9;
  int KT = Ktot >> 5;
  int nt = frag / KT, kt = frag - nt * KT;
  int n = nt * 16 + (lane & 15);
  int k = kt * 32 + ((lane >> 4) << 3) + (j & 7) + ((j >> 3) << 4);
  const float* W = (k < K1) ? W0 : W1;
  int kk = (k < K1) ? k : (k - K1);
  int Kd = (k < K1) ? K1 : (Ktot - K1);
  float v = transposed ? W[n * Kd + kk] : W[kk * Nout + n];
  out[idx] = __float2bfloat16(v);
}

// ---------------- base1/base2 precompute (single pass over X) ----------------
__global__ void base_kernel(const float* __restrict__ X, const float* __restrict__ y_prev,
                            const float* __restrict__ attn1_W, const float* __restrict__ attn1_b,
                            const float* __restrict__ attn2_W, const float* __restrict__ attn2_b,
                            float* __restrict__ base1, float* __restrict__ base2) {
  int idx = blockIdx.x * blockDim.x + threadIdx.x;
  if (idx >= BSZ * NF) return;
  int b = idx >> 7, n = idx & 127;
  const float* Wx1 = attn1_W + 256;
  const float* Wx2 = attn2_W + 256;
  const float* Wy2 = attn2_W + 256 + TM1;
  float s1 = 0.f, s2 = 0.f, sy = 0.f;
#pragma unroll
  for (int t = 0; t < TM1; ++t) {
    float x = X[((size_t)b * TM1 + t) * NF + n];
    s1 += x * Wx1[t];
    s2 += x * Wx2[t];
    sy += y_prev[b * TM1 + t] * Wy2[t];
  }
  base1[idx] = s1 + attn1_b[0];
  base2[idx] = s2 + sy + attn2_b[0];
}

// ---------------- encoder: persistent over 9 steps, 64 rows / WG ----------------
#define EMT 64
__global__ __launch_bounds__(256) void encoder_kernel(
    const float* __restrict__ X, const float* __restrict__ base1, const float* __restrict__ base2,
    const float* __restrict__ attn1_W, const float* __restrict__ attn2_W,
    const __hip_bfloat16* __restrict__ pkE1, const __hip_bfloat16* __restrict__ pkE2,
    const float* __restrict__ e1_bih, const float* __restrict__ e1_bhh,
    const float* __restrict__ e2_bih, const float* __restrict__ e2_bhh,
    __hip_bfloat16* __restrict__ X_enc) {
  extern __shared__ char smem[];
  float* g    = (float*)smem;                                  // 64x512 f32 = 128 KB
  float* xtf  = (float*)(smem + 131072);                       // 64x128 f32
  float* scr2 = (float*)(smem + 131072 + 32768);               // 64x128 f32
  __hip_bfloat16* A1 = (__hip_bfloat16*)(smem + 196608);       // [x_t | h]  64x256 bf16
  __hip_bfloat16* A2 = (__hip_bfloat16*)(smem + 196608 + 32768); // [x_t2 | hs]
  float* red1 = (float*)(smem + 262144);
  float* red2 = red1 + EMT;
  float* sum1 = red2 + EMT;
  float* sum2 = sum1 + EMT;

  const int tid  = threadIdx.x;
  const int row0 = blockIdx.x * EMT;

  float s1c[32], s2c[32];  // cell states, reg-resident across all steps
#pragma unroll
  for (int j = 0; j < 32; ++j) { s1c[j] = 0.f; s2c[j] = 0.f; }
  for (int idx = tid; idx < EMT * 128; idx += 256) {
    int r = idx >> 7, e = idx & 127;
    A1[r * 256 + 128 + e] = __float2bfloat16(0.f);
    A2[r * 256 + 128 + e] = __float2bfloat16(0.f);
  }
  __syncthreads();

  for (int t = 0; t < TM1; ++t) {
    if (tid < EMT) { red1[tid] = 0.f; red2[tid] = 0.f; sum1[tid] = 0.f; sum2[tid] = 0.f; }
    __syncthreads();
    // per-row attention scalars: h.Wh + c.Ws for both stages
#pragma unroll
    for (int j = 0; j < 32; ++j) {
      int idx = tid + 256 * j;
      int r = idx >> 7, e = idx & 127;
      float h  = __bfloat162float(A1[r * 256 + 128 + e]);
      float hs = __bfloat162float(A2[r * 256 + 128 + e]);
      atomicAdd(&red1[r], h * attn1_W[e] + s1c[j] * attn1_W[128 + e]);
      atomicAdd(&red2[r], hs * attn2_W[e] + s2c[j] * attn2_W[128 + e]);
    }
    __syncthreads();
    // softmax numerators + sums (values are small: skip max-shift)
    for (int idx = tid; idx < EMT * 128; idx += 256) {
      int r = idx >> 7, e = idx & 127;
      float v1 = __expf(base1[(size_t)(row0 + r) * 128 + e] + red1[r]);
      float v2 = __expf(base2[(size_t)(row0 + r) * 128 + e] + red2[r]);
      xtf[idx] = v1; scr2[idx] = v2;
      atomicAdd(&sum1[r], v1);
      atomicAdd(&sum2[r], v2);
    }
    __syncthreads();
    // x_t = a1 * x
    for (int idx = tid; idx < EMT * 128; idx += 256) {
      int r = idx >> 7, e = idx & 127;
      float x  = X[((size_t)(row0 + r) * TM1 + t) * NF + e];
      float xv = (xtf[idx] / sum1[r]) * x;
      xtf[idx] = xv;
      A1[r * 256 + e] = __float2bfloat16(xv);
    }
    __syncthreads();
    // gates1 = [x_t | h] @ [Wih;Whh]^T   (64x256 @ 256x512)
    gemm_tiles<8, false>(A1, pkE1, g, 4, 32, 256, 512);
    __syncthreads();
    // LSTM1 elementwise + build x_t2 = a2 * x_t
#pragma unroll
    for (int j = 0; j < 32; ++j) {
      int idx = tid + 256 * j;
      int r = idx >> 7, e = idx & 127;
      float gi = g[r * 512 + e]        + e1_bih[e]        + e1_bhh[e];
      float gf = g[r * 512 + 128 + e]  + e1_bih[128 + e]  + e1_bhh[128 + e];
      float gg = g[r * 512 + 256 + e]  + e1_bih[256 + e]  + e1_bhh[256 + e];
      float go = g[r * 512 + 384 + e]  + e1_bih[384 + e]  + e1_bhh[384 + e];
      float c2 = sigm(gf) * s1c[j] + sigm(gi) * tanhf(gg);
      s1c[j] = c2;
      A1[r * 256 + 128 + e] = __float2bfloat16(sigm(go) * tanhf(c2));
      float xt2 = (scr2[idx] / sum2[r]) * xtf[idx];
      A2[r * 256 + e] = __float2bfloat16(xt2);
    }
    __syncthreads();
    // gates2 = [x_t2 | hs] @ W2   (64x256 @ 256x512)
    gemm_tiles<8, false>(A2, pkE2, g, 4, 32, 256, 512);
    __syncthreads();
#pragma unroll
    for (int j = 0; j < 32; ++j) {
      int idx = tid + 256 * j;
      int r = idx >> 7, e = idx & 127;
      float gi = g[r * 512 + e]        + e2_bih[e]        + e2_bhh[e];
      float gf = g[r * 512 + 128 + e]  + e2_bih[128 + e]  + e2_bhh[128 + e];
      float gg = g[r * 512 + 256 + e]  + e2_bih[256 + e]  + e2_bhh[256 + e];
      float go = g[r * 512 + 384 + e]  + e2_bih[384 + e]  + e2_bhh[384 + e];
      float c2 = sigm(gf) * s2c[j] + sigm(gi) * tanhf(gg);
      s2c[j] = c2;
      float h2 = sigm(go) * tanhf(c2);
      A2[r * 256 + 128 + e] = __float2bfloat16(h2);
      X_enc[((size_t)(row0 + r) * TM1 + t) * 128 + e] = __float2bfloat16(h2);
    }
    __syncthreads();
  }
}

// ---------------- decoder: persistent over 9 steps, 32 rows / WG ----------------
#define DMT 32
__global__ __launch_bounds__(256) void decoder_kernel(
    const float* __restrict__ y_prev, const __hip_bfloat16* __restrict__ X_enc,
    const __hip_bfloat16* __restrict__ pkW1e, const __hip_bfloat16* __restrict__ pkW1dc,
    const __hip_bfloat16* __restrict__ pkWd,
    const float* __restrict__ dattn_b1, const float* __restrict__ dattn_W2,
    const float* __restrict__ dattn_b2, const float* __restrict__ dlstm_Wih,
    const float* __restrict__ dlstm_bih, const float* __restrict__ dlstm_bhh,
    const float* __restrict__ fc_W, const float* __restrict__ fc_b,
    const float* __restrict__ fcf_W, const float* __restrict__ fcf_b,
    float* __restrict__ out) {
  extern __shared__ char smem[];
  __hip_bfloat16* xe    = (__hip_bfloat16*)smem;             // 288x128 bf16 = 72 KB
  __hip_bfloat16* baseD = (__hip_bfloat16*)(smem + 73728);   // 288x128 bf16
  float* g   = (float*)(smem + 147456);                       // 32x512 f32 (gates / vd)
  __hip_bfloat16* Adc = (__hip_bfloat16*)(smem + 212992);    // [d | c] 32x256 bf16
  float* ctx  = (float*)(smem + 229376);                      // 32x128 f32
  float* beta = (float*)(smem + 245760);                      // 288
  float* ytld = beta + 288;                                   // 32
  float* ored = ytld + 32;                                    // 32

  const int tid  = threadIdx.x;
  const int row0 = blockIdx.x * DMT;

  // stage X_enc tile (contiguous), zero [d|c] and ctx
  for (int idx = tid; idx < DMT * TM1 * 128; idx += 256)
    xe[idx] = X_enc[(size_t)row0 * TM1 * 128 + idx];
  for (int idx = tid; idx < DMT * 256; idx += 256) Adc[idx] = __float2bfloat16(0.f);
  for (int idx = tid; idx < DMT * 128; idx += 256) ctx[idx] = 0.f;
  float creg[16];
#pragma unroll
  for (int j = 0; j < 16; ++j) creg[j] = 0.f;
  __syncthreads();
  // baseD = X_enc @ W1_e   (288x128 @ 128x128), bias folded in later
  gemm_tiles<4, true>(xe, pkW1e, baseD, 18, 8, 128, 128);
  __syncthreads();

  for (int t = 0; t < TM1; ++t) {
    for (int i = tid; i < 320; i += 256) { if (i < 288) beta[i] = 0.f; else ytld[i - 288] = 0.f; }
    // vd = [d|c] @ [W1_d;W1_c]   (32x256 @ 256x128) -> g[0:4096]
    gemm_tiles<8, false>(Adc, pkW1dc, g, 2, 8, 256, 128);
    __syncthreads();
    // scores: sum_f tanh(baseD + vd + b1) * W2
    for (int idx = tid; idx < 288 * 128; idx += 256) {
      int rt = idx >> 7, f = idx & 127;
      int r = rt / TM1;
      float pre = tanhf(__bfloat162float(baseD[idx]) + g[r * 128 + f] + dattn_b1[f]);
      atomicAdd(&beta[rt], pre * dattn_W2[f]);
    }
    __syncthreads();
    if (tid < DMT) {
      float e_[TM1]; float s = 0.f;
#pragma unroll
      for (int tt = 0; tt < TM1; ++tt) { e_[tt] = __expf(beta[tid * TM1 + tt] + dattn_b2[0]); s += e_[tt]; }
      float inv = 1.f / s;
#pragma unroll
      for (int tt = 0; tt < TM1; ++tt) beta[tid * TM1 + tt] = e_[tt] * inv;
    }
    __syncthreads();
    // ctx + y_tilde dot
    for (int idx = tid; idx < DMT * 128; idx += 256) {
      int r = idx >> 7, h = idx & 127;
      float acc = 0.f;
#pragma unroll
      for (int tt = 0; tt < TM1; ++tt)
        acc += beta[r * TM1 + tt] * __bfloat162float(xe[(r * TM1 + tt) * 128 + h]);
      ctx[idx] = acc;
      atomicAdd(&ytld[r], acc * fc_W[h]);
    }
    __syncthreads();
    if (tid < DMT)
      ytld[tid] += y_prev[(size_t)(row0 + tid) * TM1 + t] * fc_W[128] + fc_b[0];
    __syncthreads();
    // gates = d @ Whh^T   (32x128 @ 128x512)
    gemm_tiles<4, false>(Adc, pkWd, g, 2, 32, 256, 512);
    __syncthreads();
#pragma unroll
    for (int j = 0; j < 16; ++j) {
      int idx = tid + 256 * j;
      int r = idx >> 7, e = idx & 127;
      float yt = ytld[r];
      float gi = g[r * 512 + e]       + yt * dlstm_Wih[e]       + dlstm_bih[e]       + dlstm_bhh[e];
      float gf = g[r * 512 + 128 + e] + yt * dlstm_Wih[128 + e] + dlstm_bih[128 + e] + dlstm_bhh[128 + e];
      float gg = g[r * 512 + 256 + e] + yt * dlstm_Wih[256 + e] + dlstm_bih[256 + e] + dlstm_bhh[256 + e];
      float go = g[r * 512 + 384 + e] + yt * dlstm_Wih[384 + e] + dlstm_bih[384 + e] + dlstm_bhh[384 + e];
      float c2 = sigm(gf) * creg[j] + sigm(gi) * tanhf(gg);
      creg[j] = c2;
      Adc[r * 256 + e]       = __float2bfloat16(sigm(go) * tanhf(c2));
      Adc[r * 256 + 128 + e] = __float2bfloat16(c2);
    }
    __syncthreads();
  }
  // y_pred = [d, ctx] . fcf_W + fcf_b
  if (tid < DMT) ored[tid] = 0.f;
  __syncthreads();
  for (int idx = tid; idx < DMT * 128; idx += 256) {
    int r = idx >> 7, h = idx & 127;
    float v = __bfloat162float(Adc[r * 256 + h]) * fcf_W[h] + ctx[idx] * fcf_W[128 + h];
    atomicAdd(&ored[r], v);
  }
  __syncthreads();
  if (tid < DMT) out[row0 + tid] = ored[tid] + fcf_b[0];
}

extern "C" void kernel_launch(void* const* d_in, const int* in_sizes, int n_in,
                              void* d_out, int out_size, void* d_ws, size_t ws_size,
                              hipStream_t stream) {
  (void)in_sizes; (void)n_in; (void)out_size; (void)ws_size;
  const float* X        = (const float*)d_in[0];
  const float* y_prev   = (const float*)d_in[1];
  const float* e1_Wih   = (const float*)d_in[2];
  const float* e1_Whh   = (const float*)d_in[3];
  const float* e1_bih   = (const float*)d_in[4];
  const float* e1_bhh   = (const float*)d_in[5];
  const float* e2_Wih   = (const float*)d_in[6];
  const float* e2_Whh   = (const float*)d_in[7];
  const float* e2_bih   = (const float*)d_in[8];
  const float* e2_bhh   = (const float*)d_in[9];
  const float* attn1_W  = (const float*)d_in[10];
  const float* attn1_b  = (const float*)d_in[11];
  const float* attn2_W  = (const float*)d_in[12];
  const float* attn2_b  = (const float*)d_in[13];
  const float* dattn_W1 = (const float*)d_in[14];
  const float* dattn_b1 = (const float*)d_in[15];
  const float* dattn_W2 = (const float*)d_in[16];
  const float* dattn_b2 = (const float*)d_in[17];
  const float* dlstm_Wih = (const float*)d_in[18];
  const float* dlstm_Whh = (const float*)d_in[19];
  const float* dlstm_bih = (const float*)d_in[20];
  const float* dlstm_bhh = (const float*)d_in[21];
  const float* fc_W  = (const float*)d_in[22];
  const float* fc_b  = (const float*)d_in[23];
  const float* fcf_W = (const float*)d_in[24];
  const float* fcf_b = (const float*)d_in[25];
  float* out = (float*)d_out;

  // workspace carve
  char* ws = (char*)d_ws;
  __hip_bfloat16* pkE1   = (__hip_bfloat16*)ws; ws += 256 * 512 * 2;
  __hip_bfloat16* pkE2   = (__hip_bfloat16*)ws; ws += 256 * 512 * 2;
  __hip_bfloat16* pkWd   = (__hip_bfloat16*)ws; ws += 128 * 512 * 2;
  __hip_bfloat16* pkW1dc = (__hip_bfloat16*)ws; ws += 256 * 128 * 2;
  __hip_bfloat16* pkW1e  = (__hip_bfloat16*)ws; ws += 128 * 128 * 2;
  float* base1 = (float*)ws; ws += (size_t)BSZ * 128 * 4;
  float* base2 = (float*)ws; ws += (size_t)BSZ * 128 * 4;
  __hip_bfloat16* X_enc = (__hip_bfloat16*)ws;  // BSZ*9*128 bf16

  // pack weights into WMMA B-fragment layout (bf16)
  pack_b_kernel<<<(256 * 512 + 255) / 256, 256, 0, stream>>>(e1_Wih, e1_Whh, 128, 256, 512, 1, pkE1);
  pack_b_kernel<<<(256 * 512 + 255) / 256, 256, 0, stream>>>(e2_Wih, e2_Whh, 128, 256, 512, 1, pkE2);
  pack_b_kernel<<<(128 * 512 + 255) / 256, 256, 0, stream>>>(dlstm_Whh, dlstm_Whh, 128, 128, 512, 1, pkWd);
  pack_b_kernel<<<(256 * 128 + 255) / 256, 256, 0, stream>>>(dattn_W1, dattn_W1, 256, 256, 128, 0, pkW1dc);
  pack_b_kernel<<<(128 * 128 + 255) / 256, 256, 0, stream>>>(dattn_W1 + 256 * 128, dattn_W1 + 256 * 128,
                                                             128, 128, 128, 0, pkW1e);
  base_kernel<<<(BSZ * 128) / 256, 256, 0, stream>>>(X, y_prev, attn1_W, attn1_b, attn2_W, attn2_b,
                                                     base1, base2);
  size_t enc_smem = 262144 + 4 * EMT * 4;  // 263168 B (< 320 KB/WGP)
  encoder_kernel<<<BSZ / EMT, 256, enc_smem, stream>>>(X, base1, base2, attn1_W, attn2_W, pkE1, pkE2,
                                                       e1_bih, e1_bhh, e2_bih, e2_bhh, X_enc);
  size_t dec_smem = 245760 + (288 + 32 + 32) * 4;  // 247168 B
  decoder_kernel<<<BSZ / DMT, 256, dec_smem, stream>>>(y_prev, X_enc, pkW1e, pkW1dc, pkWd,
                                                       dattn_b1, dattn_W2, dattn_b2, dlstm_Wih,
                                                       dlstm_bih, dlstm_bhh, fc_W, fc_b, fcf_W, fcf_b,
                                                       out);
}
